// AttnDecoderRNN_7559142441063
// MI455X (gfx1250) — compile-verified, matrix operand order checked by reading, not run on hardware
//
#include <hip/hip_runtime.h>
#include <hip/hip_bf16.h>

#define Hdim 256
#define Bdim 256
#define Sdim 2048
#define Pdim 3
#define ROWS (Sdim * Bdim)   // 524288 (s,b) rows

typedef __attribute__((ext_vector_type(16))) __bf16 v16bf;
typedef __attribute__((ext_vector_type(8)))  __bf16 v8bf;
typedef __attribute__((ext_vector_type(8)))  float  v8f;

__device__ inline __bf16 f2bf(float f) {
    unsigned u = __builtin_bit_cast(unsigned, f);
    unsigned r = u + 0x7FFFu + ((u >> 16) & 1u);   // round-to-nearest-even
    unsigned short h = (unsigned short)(r >> 16);
    return __builtin_bit_cast(__bf16, h);
}
__device__ inline float bf2f(__bf16 b) {
    unsigned short h = __builtin_bit_cast(unsigned short, b);
    unsigned u = ((unsigned)h) << 16;
    return __builtin_bit_cast(float, u);
}

// ---------------- K1: hid_[b,n] = dot(h_prev[b,:], Wh[n,:]) + bh[n] ----------------
__global__ void __launch_bounds__(256) hid_proj(const float* __restrict__ hprev,
                                                const float* __restrict__ Wh,
                                                const float* __restrict__ bh,
                                                float* __restrict__ hid) {
    int b = blockIdx.x, n = threadIdx.x;
    const float* x = hprev + b * Hdim;
    const float* w = Wh + n * Hdim;
    float acc = bh[n];
    #pragma unroll 8
    for (int k = 0; k < Hdim; ++k) acc = fmaf(x[k], w[k], acc);
    hid[b * Hdim + n] = acc;
}

// ---------------- K2: fused  energy[s,b] = Wv . sigmoid(enc@We^T + be + hid_[b]) ----
// bf16 WMMA 16x16x32, f32 accumulate. Block = 256 thr = 8 waves, slab = 128 rows.
// LDS: swizzled We bf16 (128KB) + hid slab bf16 (64KB) + A-chunk stage (8KB) = 200KB.
__global__ void __launch_bounds__(256) fused_attn_energy(
    const float* __restrict__ enc, const float* __restrict__ We,
    const float* __restrict__ be,  const float* __restrict__ Wv,
    const float* __restrict__ hid, float* __restrict__ energy)
{
    extern __shared__ char smem[];
    __bf16* sWe  = (__bf16*)smem;                       // [16 nt][8 c][32 lane][16 t]
    __bf16* sHid = (__bf16*)(smem + 131072);            // [128 m][256 n]
    __bf16* sA   = (__bf16*)(smem + 131072 + 65536);    // [128 m][32 k]

    const int tid  = threadIdx.x;
    const int lane = tid & 31;
    const int wave = tid >> 5;
    const int grp  = lane >> 4;   // which 16-lane half
    const int ln   = lane & 15;

    // Stage We once per block, pre-swizzled into the exact WMMA B-operand
    // per-lane order (mirrors the 16-bit A layout: runs of 8 K at grp*8 and grp*8+16).
    for (int idx = tid; idx < 16 * 8 * 32 * 16; idx += 256) {
        int t  = idx & 15;
        int l  = (idx >> 4) & 31;
        int c  = (idx >> 9) & 7;
        int nt = idx >> 12;
        int n  = nt * 16 + (l & 15);
        int kl = (t < 8 ? t : t + 8) + (l >> 4) * 8;
        int k  = c * 32 + kl;
        sWe[idx] = f2bf(We[n * Hdim + k]);
    }

    const int nslabs = ROWS / 128;   // 4096
    for (int slab = blockIdx.x; slab < nslabs; slab += gridDim.x) {
        const int r0 = slab * 128;
        __syncthreads();   // protect sHid/sA from previous slab's readers

        // hid rows for this slab: b = (r0+m) & 255  (contiguous mod B)
        for (int idx = tid; idx < 128 * 256; idx += 256) {
            int m = idx >> 8, n = idx & 255;
            int b = (r0 + m) & (Bdim - 1);
            sHid[idx] = f2bf(hid[b * Hdim + n]);
        }

        // Preload this wave's 16x256 A panel through LDS, chunk by chunk.
        v16bf aw[8];
        #pragma unroll
        for (int c = 0; c < 8; ++c) {
            __syncthreads();
            for (int idx = tid; idx < 128 * 32; idx += 256) {
                int m = idx >> 5, kk = idx & 31;
                sA[idx] = f2bf(enc[(size_t)(r0 + m) * Hdim + c * 32 + kk]);
            }
            __syncthreads();
            const __bf16* base = &sA[(wave * 16 + ln) * 32 + grp * 8];
            v8bf lo = *(const v8bf*)(base);
            v8bf hi = *(const v8bf*)(base + 16);
            aw[c] = __builtin_shufflevector(lo, hi,
                     0,1,2,3,4,5,6,7,8,9,10,11,12,13,14,15);
        }

        float erow[8] = {0,0,0,0,0,0,0,0};   // per-row energy partials

        #pragma unroll
        for (int nt = 0; nt < 16; ++nt) {
            v8f acc = {};
            #pragma unroll
            for (int c = 0; c < 8; ++c) {
                const __bf16* bb = &sWe[((nt * 8 + c) * 32 + lane) * 16];
                v8bf blo = *(const v8bf*)(bb);
                v8bf bhi = *(const v8bf*)(bb + 8);
                v16bf bv = __builtin_shufflevector(blo, bhi,
                            0,1,2,3,4,5,6,7,8,9,10,11,12,13,14,15);
                acc = __builtin_amdgcn_wmma_f32_16x16x32_bf16(
                        false, aw[c], false, bv, (short)0, acc, false, false);
            }
            // epilogue: + be + hid, sigmoid, dot with Wv (fused, enc_ never stored)
            int   n   = nt * 16 + ln;
            float ben = be[n];
            float wvn = Wv[n];
            #pragma unroll
            for (int v = 0; v < 8; ++v) {
                int   srow = wave * 16 + v + 8 * grp;       // C layout: M = v + 8*grp
                float hv   = bf2f(sHid[srow * 256 + n]);
                float val  = acc[v] + ben + hv;
                float sg   = 1.0f / (1.0f + __expf(-val));
                erow[v]    = fmaf(wvn, sg, erow[v]);
            }
        }

        // reduce the 16 N-lanes of each half -> one energy value per row
        #pragma unroll
        for (int v = 0; v < 8; ++v) {
            float e = erow[v];
            e += __shfl_xor(e, 1, 32);
            e += __shfl_xor(e, 2, 32);
            e += __shfl_xor(e, 4, 32);
            e += __shfl_xor(e, 8, 32);
            if (ln == 0) {
                int r = r0 + wave * 16 + v + 8 * grp;
                energy[r] = e;
            }
        }
    }
}

// ---------------- K3: softmax over s (per column b); writes attn [b][s] ----------
__global__ void __launch_bounds__(256) softmax_seq(const float* __restrict__ energy,
                                                   float* __restrict__ attn) {
    int b = blockIdx.x, t = threadIdx.x;
    float ev[Sdim / 256];
    float mx = -3.0e38f;
    #pragma unroll
    for (int i = 0; i < Sdim / 256; ++i) {
        int s = t + i * 256;
        ev[i] = energy[s * Bdim + b];
        mx = fmaxf(mx, ev[i]);
    }
    __shared__ float red[256];
    red[t] = mx; __syncthreads();
    for (int off = 128; off > 0; off >>= 1) {
        if (t < off) red[t] = fmaxf(red[t], red[t + off]);
        __syncthreads();
    }
    mx = red[0]; __syncthreads();
    float sum = 0.0f;
    #pragma unroll
    for (int i = 0; i < Sdim / 256; ++i) { ev[i] = __expf(ev[i] - mx); sum += ev[i]; }
    red[t] = sum; __syncthreads();
    for (int off = 128; off > 0; off >>= 1) {
        if (t < off) red[t] += red[t + off];
        __syncthreads();
    }
    float inv = 1.0f / red[0];
    #pragma unroll
    for (int i = 0; i < Sdim / 256; ++i) attn[b * Sdim + t + i * 256] = ev[i] * inv;
}

// ---------------- K4: context partials (deterministic, no float atomics) ---------
__global__ void __launch_bounds__(256) ctx_partial(const float* __restrict__ attn,
                                                   const float* __restrict__ enc,
                                                   float* __restrict__ cpart) {
    int b = blockIdx.x, ch = blockIdx.y, h = threadIdx.x;
    int s0 = ch * (Sdim / 8);
    float acc = 0.0f;
    #pragma unroll 4
    for (int i = 0; i < Sdim / 8; ++i) {
        int s = s0 + i;
        acc = fmaf(attn[b * Sdim + s], enc[((size_t)s * Bdim + b) * Hdim + h], acc);
    }
    cpart[((size_t)ch * Bdim + b) * Hdim + h] = acc;
}

__global__ void __launch_bounds__(256) ctx_reduce(const float* __restrict__ cpart,
                                                  float* __restrict__ ctx) {
    int i = blockIdx.x * 256 + threadIdx.x;
    float acc = 0.0f;
    #pragma unroll
    for (int c = 0; c < 8; ++c) acc += cpart[(size_t)c * Bdim * Hdim + i];
    ctx[i] = acc;
}

// ---------------- K5: GRU cell --------------------------------------------------
__global__ void __launch_bounds__(256) gru_cell(
    const float* __restrict__ pal, const float* __restrict__ ctx,
    const float* __restrict__ hprev,
    const float* __restrict__ W_ih, const float* __restrict__ b_ih,
    const float* __restrict__ W_hh, const float* __restrict__ b_hh,
    float* __restrict__ hout)
{
    int b = blockIdx.x, h = threadIdx.x;
    const float* x1 = pal + b * Pdim;
    const float* x2 = ctx + b * Hdim;
    const float* hp = hprev + b * Hdim;
    float gi[3], gh[3];
    #pragma unroll
    for (int g = 0; g < 3; ++g) {
        int row = g * Hdim + h;
        const float* wi = W_ih + (size_t)row * (Hdim + Pdim);
        float a = b_ih[row];
        a = fmaf(x1[0], wi[0], a);
        a = fmaf(x1[1], wi[1], a);
        a = fmaf(x1[2], wi[2], a);
        #pragma unroll 8
        for (int k = 0; k < Hdim; ++k) a = fmaf(x2[k], wi[Pdim + k], a);
        gi[g] = a;
        const float* wh = W_hh + (size_t)row * Hdim;
        float c = b_hh[row];
        #pragma unroll 8
        for (int k = 0; k < Hdim; ++k) c = fmaf(hp[k], wh[k], c);
        gh[g] = c;
    }
    float r = 1.0f / (1.0f + __expf(-(gi[0] + gh[0])));
    float z = 1.0f / (1.0f + __expf(-(gi[1] + gh[1])));
    float n = tanhf(gi[2] + r * gh[2]);
    hout[b * Hdim + h] = (1.0f - z) * n + z * hp[h];
}

// ---------------- K6: h1 = relu(gru @ W1^T + b1) --------------------------------
__global__ void __launch_bounds__(256) mlp1(const float* __restrict__ hin,
                                            const float* __restrict__ W1,
                                            const float* __restrict__ b1,
                                            float* __restrict__ h1) {
    int b = blockIdx.x, h = threadIdx.x;
    const float* x = hin + b * Hdim;
    const float* w = W1 + h * Hdim;
    float a = b1[h];
    #pragma unroll 8
    for (int k = 0; k < Hdim; ++k) a = fmaf(x[k], w[k], a);
    h1[b * Hdim + h] = fmaxf(a, 0.0f);
}

// ---------------- K7: BN stats (biased var, training mode) ----------------------
__global__ void __launch_bounds__(256) bn_stats(const float* __restrict__ h1,
                                                float* __restrict__ mu,
                                                float* __restrict__ var) {
    int h = blockIdx.x, b = threadIdx.x;
    float v = h1[b * Hdim + h];
    __shared__ float rs[256], rq[256];
    rs[b] = v; rq[b] = v * v; __syncthreads();
    for (int off = 128; off > 0; off >>= 1) {
        if (b < off) { rs[b] += rs[b + off]; rq[b] += rq[b + off]; }
        __syncthreads();
    }
    if (b == 0) {
        float m = rs[0] * (1.0f / Bdim);
        mu[h]  = m;
        var[h] = rq[0] * (1.0f / Bdim) - m * m;
    }
}

// ---------------- K8: palette = BN(h1) @ W2^T + b2 ------------------------------
__global__ void __launch_bounds__(256) mlp2(const float* __restrict__ h1,
                                            const float* __restrict__ mu,
                                            const float* __restrict__ var,
                                            const float* __restrict__ gamma,
                                            const float* __restrict__ beta,
                                            const float* __restrict__ W2,
                                            const float* __restrict__ b2,
                                            float* __restrict__ pal) {
    int b = blockIdx.x, p = blockIdx.y, h = threadIdx.x;
    float x  = h1[b * Hdim + h];
    float xn = (x - mu[h]) * rsqrtf(var[h] + 1e-5f) * gamma[h] + beta[h];
    float t  = xn * W2[p * Hdim + h];
    __shared__ float rs[256];
    rs[h] = t; __syncthreads();
    for (int off = 128; off > 0; off >>= 1) {
        if (h < off) rs[h] += rs[h + off];
        __syncthreads();
    }
    if (h == 0) pal[b * Pdim + p] = rs[0] + b2[p];
}

extern "C" void kernel_launch(void* const* d_in, const int* in_sizes, int n_in,
                              void* d_out, int out_size, void* d_ws, size_t ws_size,
                              hipStream_t stream) {
    (void)in_sizes; (void)n_in; (void)out_size; (void)ws_size;
    const float* last_palette = (const float*)d_in[0];
    const float* hprev        = (const float*)d_in[1];
    const float* enc          = (const float*)d_in[2];
    const float* We   = (const float*)d_in[3];
    const float* be   = (const float*)d_in[4];
    const float* Wh   = (const float*)d_in[5];
    const float* bh   = (const float*)d_in[6];
    const float* Wv   = (const float*)d_in[7];
    /* d_in[8] = bv: softmax is shift-invariant and energy is not an output -> unused */
    const float* W_ih = (const float*)d_in[9];
    const float* b_ih = (const float*)d_in[10];
    const float* W_hh = (const float*)d_in[11];
    const float* b_hh = (const float*)d_in[12];
    const float* W1   = (const float*)d_in[13];
    const float* b1   = (const float*)d_in[14];
    const float* gam  = (const float*)d_in[15];
    const float* bet  = (const float*)d_in[16];
    const float* W2   = (const float*)d_in[17];
    const float* b2   = (const float*)d_in[18];

    float* out = (float*)d_out;
    float* out_pal  = out;                        // [B,3]
    float* out_ctx  = out + Bdim * Pdim;          // [1,B,1,H]
    float* out_gru  = out_ctx + Bdim * Hdim;      // [B,H]
    float* out_attn = out_gru + Bdim * Hdim;      // [B,1,S]

    float* ws       = (float*)d_ws;
    float* w_hid    = ws;                           // B*H
    float* w_energy = ws + Bdim * Hdim;             // S*B
    float* w_cpart  = w_energy;                     // 8*B*H, aliases energy (freed by K3)
    float* w_h1     = w_energy + (size_t)Sdim * Bdim; // B*H
    float* w_mu     = w_h1 + Bdim * Hdim;           // H
    float* w_var    = w_mu + Hdim;                  // H

    const int smemK2 = 131072 + 65536 + 8192;       // 200 KB < 320 KB/WGP
    hipFuncSetAttribute((const void*)fused_attn_energy,
                        hipFuncAttributeMaxDynamicSharedMemorySize, smemK2);

    hid_proj<<<Bdim, 256, 0, stream>>>(hprev, Wh, bh, w_hid);
    fused_attn_energy<<<1024, 256, smemK2, stream>>>(enc, We, be, Wv, w_hid, w_energy);
    softmax_seq<<<Bdim, 256, 0, stream>>>(w_energy, out_attn);
    ctx_partial<<<dim3(Bdim, 8), 256, 0, stream>>>(out_attn, enc, w_cpart);
    ctx_reduce<<<Bdim, 256, 0, stream>>>(w_cpart, out_ctx);
    gru_cell<<<Bdim, 256, 0, stream>>>(last_palette, out_ctx, hprev,
                                       W_ih, b_ih, W_hh, b_hh, out_gru);
    mlp1<<<Bdim, 256, 0, stream>>>(out_gru, W1, b1, w_h1);
    bn_stats<<<Hdim, 256, 0, stream>>>(w_h1, w_mu, w_var);
    mlp2<<<dim3(Bdim, Pdim), 256, 0, stream>>>(w_h1, w_mu, w_var, gam, bet, W2, b2, out_pal);
}
